// TabDPTModel_89593017795265
// MI455X (gfx1250) — matrix operand order repaired and checked by reading, not compiled, not run on hardware
//
#include <hip/hip_runtime.h>
#include <hip/hip_bf16.h>
#include <math.h>

// Problem constants (from reference)
#define DD    1024
#define HH    8
#define HDD   128
#define NLL   3
#define NFF   100
#define NHIDD 4096
#define NOUTT 10
#define BB    4
#define LL    2048
#define PPP   1024
#define KENC  128            // NF padded to WMMA K multiple
#define MTOK  (BB*LL)        // 8192 tokens
#define CLIPS 4.0f

typedef __attribute__((ext_vector_type(16))) __bf16 v16bf;
typedef __attribute__((ext_vector_type(8)))  __bf16 v8bf;
typedef __attribute__((ext_vector_type(8)))  float  v8f;

// ---------------------------------------------------------------------------
// WMMA fragment helpers (CDNA5 wave32, V_WMMA_F32_16X16X32_BF16)
// A 16x32 bf16: lanes 0-15 row M=lane hold K=[0..15] (v0-7); lanes 16-31
// hold K=[16..31] (contiguous, same k order as B).
// B 32x16 bf16: lane holds column N=lane%16, 16 contiguous K values starting
// at (lane/16)*16.
// C/D 16x16 f32: lane holds (m = i + 8*(lane>>4), n = lane&15) in vgpr i.
// ---------------------------------------------------------------------------
__device__ __forceinline__ v16bf load_a_frag(const __bf16* rowPtr, int lhi) {
    const v8bf a0 = *(const v8bf*)(rowPtr + lhi * 16);
    const v8bf a1 = *(const v8bf*)(rowPtr + lhi * 16 + 8);
    v16bf r;
#pragma unroll
    for (int i = 0; i < 8; ++i) { r[i] = a0[i]; r[i + 8] = a1[i]; }
    return r;
}

__device__ __forceinline__ float gelu_exact(float v) {
    return 0.5f * v * (1.0f + erff(v * 0.70710678118654752f));
}

// gfx1250 async global->LDS copy (16B per lane) + ASYNCcnt waits.
// Flat address of a __shared__ object truncates to its LDS byte offset.
__device__ __forceinline__ void async_copy_b128(unsigned ldsOff, const void* gsrc) {
    asm volatile("global_load_async_to_lds_b128 %0, %1, off"
                 :: "v"(ldsOff), "v"((unsigned long long)gsrc)
                 : "memory");
}
__device__ __forceinline__ void wait_async0() {
    asm volatile("s_wait_asynccnt 0x0" ::: "memory");
}
__device__ __forceinline__ void wait_async2() {   // allow 2 in flight
    asm volatile("s_wait_asynccnt 0x2" ::: "memory");
}

// ---------------------------------------------------------------------------
// Unified NT GEMM:  C[M,N] = A[M,K](bf16,row) x Bt[N,K](bf16,row)^T
// (+bias)(gelu)(+residual).  Block = 256 thr = 8 waves; tile 128x128; each
// wave owns a 16-row strip with 8 16x16 accumulators.  The 128x32 B tile
// (8KB) is double-buffered in LDS and staged with global_load_async_to_lds_
// b128 (2 per thread), issued one iteration ahead so the copy overlaps the
// WMMA work; s_wait_asynccnt 2 only blocks on the previous tile.  Fragments
// are contiguous ds_load_b128 pairs (40-half row pad => conflict-free).
// Batched via grid.z; C batch offset is (z/hdiv)*cOuter + (z%hdiv)*cInner
// to support (b,h)-split outputs.
// ---------------------------------------------------------------------------
template<bool BIAS, bool GELU, bool RESID, bool OUTBF16>
__global__ __launch_bounds__(256)
void gemm_bt(const __bf16* __restrict__ A, const __bf16* __restrict__ Bt,
             const float* __restrict__ bias,
             float* __restrict__ Cf, __bf16* __restrict__ Cb,
             int M, int N, int K, int ldc,
             long aBatch, long bBatch, long cOuter, long cInner, int hdiv)
{
    __shared__ __bf16 Bs[2][128][40];  // [buf][n_local][k_local], 80B rows
    const int tid  = threadIdx.x;
    const int wave = tid >> 5;
    const int lane = tid & 31;
    const int lhi  = lane >> 4;
    const int llo  = lane & 15;
    const int z    = blockIdx.z;

    const long aOff = (long)z * aBatch;
    const long bOff = (long)z * bBatch;
    const long cOff = (long)(z / hdiv) * cOuter + (long)(z % hdiv) * cInner;
    const int  m0   = blockIdx.y * 128 + wave * 16;
    const int  n0   = blockIdx.x * 128;

    v8f acc[8];
#pragma unroll
    for (int t = 0; t < 8; ++t)
#pragma unroll
        for (int i = 0; i < 8; ++i) acc[t][i] = 0.0f;

    const __bf16* Arow = A + aOff + (long)(m0 + llo) * K;

    // Async staging coords: 128x32-half tile, 32B (2 x b128) per thread.
    const int nr = tid >> 1;            // 0..127
    const int kc = (tid & 1) * 16;      // 0 or 16
    const __bf16* Bsrc = Bt + bOff + (long)(n0 + nr) * K + kc;
    const unsigned lds0 = (unsigned)(unsigned long long)(const void*)&Bs[0][nr][kc];
    const unsigned lds1 = (unsigned)(unsigned long long)(const void*)&Bs[1][nr][kc];

    // Prologue: stage tile 0 into buffer 0.
    async_copy_b128(lds0,      Bsrc);
    async_copy_b128(lds0 + 16, Bsrc + 8);

    int cur = 0;
    for (int k0 = 0; k0 < K; k0 += 32) {
        const bool more = (k0 + 32 < K);
        if (more) {                     // stage next tile into the other buffer
            const unsigned d = cur ? lds0 : lds1;
            async_copy_b128(d,      Bsrc + k0 + 32);
            async_copy_b128(d + 16, Bsrc + k0 + 40);
            __builtin_prefetch(Arow + k0 + 32, 0, 1);  // global_prefetch_b8
            wait_async2();              // previous tile landed; next in flight
        } else {
            wait_async0();
        }
        __syncthreads();                // all waves' copies visible

        const v16bf af = load_a_frag(Arow + k0, lhi);
#pragma unroll
        for (int ct = 0; ct < 8; ++ct) {
            v16bf bf_ = *(const v16bf*)(&Bs[cur][ct * 16 + llo][lhi * 16]);
            acc[ct] = __builtin_amdgcn_wmma_f32_16x16x32_bf16(
                false, af, false, bf_, (short)0, acc[ct], false, false);
        }
        __syncthreads();                // reads done before this buf restaged
        cur ^= 1;
    }

#pragma unroll
    for (int ct = 0; ct < 8; ++ct) {
        const int gn = n0 + ct * 16 + llo;
        if (gn >= N) continue;
        const float bv = BIAS ? bias[gn] : 0.0f;
#pragma unroll
        for (int i = 0; i < 8; ++i) {
            const int  gm   = m0 + lhi * 8 + i;
            const long cidx = cOff + (long)gm * ldc + gn;
            float v = acc[ct][i];
            if (BIAS) v += bv;
            if (GELU) v = gelu_exact(v);
            if (OUTBF16) {
                Cb[cidx] = (__bf16)v;
            } else {
                if (RESID) v += Cf[cidx];
                Cf[cidx] = v;
            }
        }
    }
}

// ---------------------------------------------------------------------------
// Block reductions
// ---------------------------------------------------------------------------
__device__ __forceinline__ void reduce2_256(float& a, float& b, float* s1, float* s2) {
    const int tid = threadIdx.x;
    s1[tid] = a; s2[tid] = b; __syncthreads();
    for (int off = 128; off > 0; off >>= 1) {
        if (tid < off) { s1[tid] += s1[tid + off]; s2[tid] += s2[tid + off]; }
        __syncthreads();
    }
    a = s1[0]; b = s2[0]; __syncthreads();
}

// ---------------------------------------------------------------------------
// Preprocessing: clip(4sigma) -> normalize -> clip(4sigma) -> nan_to_num
// using stats over the first P rows, per (batch, feature) column.  Writes
// the encoder A matrix xp[(l*B+b), KENC] in bf16.
// ---------------------------------------------------------------------------
__global__ __launch_bounds__(256)
void preprocess_kernel(const float* __restrict__ x_src, __bf16* __restrict__ xp)
{
    __shared__ float s1[256], s2[256];
    const int bf = blockIdx.x;
    const int b  = bf / NFF;
    const int f  = bf % NFF;
    const int tid = threadIdx.x;
    const float* col = x_src + (long)b * LL * NFF + f;

    float sum = 0.f, ss = 0.f;
    for (int l = tid; l < PPP; l += 256) { float v = col[(long)l * NFF]; sum += v; ss += v * v; }
    reduce2_256(sum, ss, s1, s2);
    const float m1  = sum / PPP;
    const float sd1 = sqrtf(fmaxf(ss / PPP - m1 * m1, 0.f));
    const float lo1 = m1 - CLIPS * sd1, hi1 = m1 + CLIPS * sd1;

    sum = 0.f; ss = 0.f;
    for (int l = tid; l < PPP; l += 256) {
        float v = fminf(fmaxf(col[(long)l * NFF], lo1), hi1);
        sum += v; ss += v * v;
    }
    reduce2_256(sum, ss, s1, s2);
    const float m2   = sum / PPP;
    const float sd2  = sqrtf(fmaxf(ss / PPP - m2 * m2, 0.f));
    const float inv2 = 1.0f / (sd2 + 1e-6f);

    sum = 0.f; ss = 0.f;
    for (int l = tid; l < PPP; l += 256) {
        float v = (fminf(fmaxf(col[(long)l * NFF], lo1), hi1) - m2) * inv2;
        sum += v; ss += v * v;
    }
    reduce2_256(sum, ss, s1, s2);
    const float m3  = sum / PPP;
    const float sd3 = sqrtf(fmaxf(ss / PPP - m3 * m3, 0.f));
    const float lo3 = m3 - CLIPS * sd3, hi3 = m3 + CLIPS * sd3;

    for (int l = tid; l < LL; l += 256) {
        float v = col[(long)l * NFF];
        v = fminf(fmaxf(v, lo1), hi1);
        v = (v - m2) * inv2;
        v = fminf(fmaxf(v, lo3), hi3);
        if (!isfinite(v)) v = 0.f;
        xp[((long)l * BB + b) * KENC + f] = (__bf16)v;
    }
}

__global__ void zero_xpad(__bf16* __restrict__ xp)
{
    long i = (long)blockIdx.x * 256 + threadIdx.x;
    const long n = (long)MTOK * (KENC - NFF);
    if (i >= n) return;
    const long r = i / (KENC - NFF);
    const int  f = NFF + (int)(i % (KENC - NFF));
    xp[r * KENC + f] = (__bf16)0.0f;
}

// Encoder epilogue: RMS-normalize, add y-encoding for l < P, relayout
// (l*B+b) -> src[(b*L+l), D] (f32).
__global__ __launch_bounds__(256)
void enc_epilogue(const float* __restrict__ xe, const float* __restrict__ y_src,
                  const float* __restrict__ yW, const float* __restrict__ yb,
                  float* __restrict__ src)
{
    __shared__ float s1[256];
    const int t = blockIdx.x;       // t = l*B + b
    const int l = t / BB, b = t % BB;
    const int tid = threadIdx.x;
    const float* row = xe + (long)t * DD;

    float ss = 0.f;
    for (int d = tid; d < DD; d += 256) { float v = row[d]; ss += v * v; }
    s1[tid] = ss; __syncthreads();
    for (int off = 128; off > 0; off >>= 1) {
        if (tid < off) s1[tid] += s1[tid + off];
        __syncthreads();
    }
    const float rinv = rsqrtf(s1[0] / DD);

    const bool addy = (l < PPP);
    const float yv = addy ? y_src[(long)b * PPP + l] : 0.f;
    float* orow = src + ((long)b * LL + l) * DD;
    for (int d = tid; d < DD; d += 256) {
        float v = row[d] * rinv;
        if (addy) v += yv * yW[d] + yb[d];
        orow[d] = v;
    }
}

// Full-width LayerNorm over D=1024, f32 in -> bf16 out.
__global__ __launch_bounds__(256)
void row_ln(const float* __restrict__ in, const float* __restrict__ g,
            const float* __restrict__ bta, __bf16* __restrict__ out)
{
    __shared__ float s1[256], s2[256];
    const int r = blockIdx.x;
    const int tid = threadIdx.x;
    const float* row = in + (long)r * DD;
    float sum = 0.f, ss = 0.f;
    for (int d = tid; d < DD; d += 256) { float v = row[d]; sum += v; ss += v * v; }
    reduce2_256(sum, ss, s1, s2);
    const float m = sum / DD;
    const float rstd = rsqrtf(fmaxf(ss / DD - m * m, 0.f) + 1e-5f);
    __bf16* orow = out + (long)r * DD;
    for (int d = tid; d < DD; d += 256)
        orow[d] = (__bf16)((row[d] - m) * rstd * g[d] + bta[d]);
}

// Per-head LayerNorm over HD=128 -> head-major bf16
// out[((bi*H + h)*rows + r)*HD + d]; input row stride configurable.
__global__ __launch_bounds__(128)
void head_ln(const float* __restrict__ in, long rowStride, int rows,
             const float* __restrict__ g, const float* __restrict__ bta,
             __bf16* __restrict__ out)
{
    __shared__ float s1[128], s2[128];
    const int idx = blockIdx.x;
    const int h  = idx % HH;
    const int r  = (idx / HH) % rows;
    const int bi = idx / (HH * rows);
    const int d  = threadIdx.x;

    const float v = in[((long)bi * rows + r) * rowStride + h * HDD + d];
    s1[d] = v; s2[d] = v * v; __syncthreads();
    for (int off = 64; off > 0; off >>= 1) {
        if (d < off) { s1[d] += s1[d + off]; s2[d] += s2[d + off]; }
        __syncthreads();
    }
    const float m    = s1[0] / HDD;
    const float rstd = rsqrtf(fmaxf(s2[0] / HDD - m * m, 0.f) + 1e-5f);
    out[(((long)bi * HH + h) * rows + r) * HDD + d] = (__bf16)((v - m) * rstd * g[d] + bta[d]);
}

// Extract V half of kv buffer, transposed per batch-head:
// vhT[((b*H+h)*HD + d)*P + s]  (the (N,K) B operand of the PV GEMM).
__global__ void extract_vT(const float* __restrict__ kvf, __bf16* __restrict__ vhT)
{
    long o = (long)blockIdx.x * 256 + threadIdx.x;
    const long n = (long)BB * HH * HDD * PPP;
    if (o >= n) return;
    const int  s    = (int)(o % PPP);
    const long rest = o / PPP;
    const int  d    = (int)(rest % HDD);
    const long bh   = rest / HDD;
    const int  h    = (int)(bh % HH);
    const int  b    = (int)(bh / HH);
    vhT[o] = (__bf16)kvf[((long)b * PPP + s) * (2 * DD) + DD + h * HDD + d];
}

// Softmax over S=1024 keys (scaled), f32 logits -> bf16 probs.
__global__ __launch_bounds__(256)
void softmax_row(const float* __restrict__ logits, __bf16* __restrict__ probs)
{
    __shared__ float s1[256];
    const long r = blockIdx.x;
    const int tid = threadIdx.x;
    const float scale = 0.08838834764831845f;  // HD^-0.5
    const float* row = logits + r * (long)PPP;

    float mx = -INFINITY;
    for (int i = tid; i < PPP; i += 256) mx = fmaxf(mx, row[i] * scale);
    s1[tid] = mx; __syncthreads();
    for (int off = 128; off > 0; off >>= 1) {
        if (tid < off) s1[tid] = fmaxf(s1[tid], s1[tid + off]);
        __syncthreads();
    }
    mx = s1[0]; __syncthreads();

    float sum = 0.f;
    for (int i = tid; i < PPP; i += 256) sum += expf(row[i] * scale - mx);
    s1[tid] = sum; __syncthreads();
    for (int off = 128; off > 0; off >>= 1) {
        if (tid < off) s1[tid] += s1[tid + off];
        __syncthreads();
    }
    const float inv = 1.0f / s1[0];

    __bf16* orow = probs + r * (long)PPP;
    for (int i = tid; i < PPP; i += 256)
        orow[i] = (__bf16)(expf(row[i] * scale - mx) * inv);
}

__global__ void f32_to_bf16(const float* __restrict__ in, __bf16* __restrict__ out, long n)
{
    long i = (long)blockIdx.x * 256 + threadIdx.x;
    if (i < n) out[i] = (__bf16)in[i];
}

// Transpose-convert a weight: in (Kin x N) f32 row-major ->
// out (Npad x Kout) bf16 row-major, zero-padded in both dims.
__global__ void convT(const float* __restrict__ in, __bf16* __restrict__ out,
                      int Kout, int Kin, int N, int Npad)
{
    long i = (long)blockIdx.x * 256 + threadIdx.x;
    const long tot = (long)Npad * Kout;
    if (i >= tot) return;
    const int n = (int)(i / Kout);
    const int k = (int)(i % Kout);
    const float v = (n < N && k < Kin) ? in[(long)k * N + n] : 0.f;
    out[i] = (__bf16)v;
}

// Final slice: pred[(b*1024+l), 11] -> out[(l, b, NOUT)] f32.
__global__ void write_out(const float* __restrict__ pred, float* __restrict__ out)
{
    long i = (long)blockIdx.x * 256 + threadIdx.x;
    const long n = (long)PPP * BB * NOUTT;
    if (i >= n) return;
    const int l = (int)(i / (BB * NOUTT));
    const int rem = (int)(i % (BB * NOUTT));
    const int b = rem / NOUTT, j = rem % NOUTT;
    out[i] = pred[((long)b * PPP + l) * (NOUTT + 1) + j];
}

// ---------------------------------------------------------------------------
// Orchestration
// ---------------------------------------------------------------------------
extern "C" void kernel_launch(void* const* d_in, const int* in_sizes, int n_in,
                              void* d_out, int out_size, void* d_ws, size_t ws_size,
                              hipStream_t stream)
{
    const float* x_src = (const float*)d_in[0];
    const float* y_src = (const float*)d_in[1];
    const float* enc_W = (const float*)d_in[2];
    const float* yenc_W = (const float*)d_in[4];
    const float* yenc_b = (const float*)d_in[5];
    const float* Wq   = (const float*)d_in[6];
    const float* Wkv  = (const float*)d_in[7];
    const float* Wout = (const float*)d_in[8];
    const float* an_g = (const float*)d_in[9];
    const float* an_b = (const float*)d_in[10];
    const float* qn_g = (const float*)d_in[11];
    const float* qn_b = (const float*)d_in[12];
    const float* kn_g = (const float*)d_in[13];
    const float* kn_b = (const float*)d_in[14];
    const float* fn_g = (const float*)d_in[15];
    const float* fn_b = (const float*)d_in[16];
    const float* ffW1 = (const float*)d_in[17];
    const float* ffb1 = (const float*)d_in[18];
    const float* ffW2 = (const float*)d_in[19];
    const float* ffb2 = (const float*)d_in[20];
    const float* hW1  = (const float*)d_in[21];
    const float* hb1  = (const float*)d_in[22];
    const float* hW2  = (const float*)d_in[23];
    const float* hb2  = (const float*)d_in[24];
    float* out = (float*)d_out;

    // -------- workspace layout --------
    char* ws = (char*)d_ws;
    size_t off = 0;
    auto alloc = [&](size_t bytes) -> void* {
        void* p = ws + off;
        off += (bytes + 255) & ~(size_t)255;
        return p;
    };
    // All weights transposed to (N, K) bf16 for the NT GEMM.
    __bf16* wqT   = (__bf16*)alloc((size_t)NLL * DD * DD * 2);
    __bf16* wkvT  = (__bf16*)alloc((size_t)NLL * 2 * DD * DD * 2);
    __bf16* woutT = (__bf16*)alloc((size_t)NLL * DD * DD * 2);
    __bf16* ffw1T = (__bf16*)alloc((size_t)NLL * NHIDD * DD * 2);
    __bf16* ffw2T = (__bf16*)alloc((size_t)NLL * DD * NHIDD * 2);
    __bf16* hw1T  = (__bf16*)alloc((size_t)NHIDD * DD * 2);
    __bf16* hw2T  = (__bf16*)alloc((size_t)128 * NHIDD * 2);  // 11 rows + pad
    __bf16* encwT = (__bf16*)alloc((size_t)DD * KENC * 2);
    __bf16* xp    = (__bf16*)alloc((size_t)MTOK * KENC * 2);
    float*  qf    = (float*) alloc((size_t)MTOK * DD * 4);    // also encoder xe
    float*  src   = (float*) alloc((size_t)MTOK * DD * 4);
    __bf16* h     = (__bf16*)alloc((size_t)MTOK * DD * 2);
    float*  kvf   = (float*) alloc((size_t)BB * PPP * 2 * DD * 4);
    __bf16* qh    = (__bf16*)alloc((size_t)BB * HH * LL * HDD * 2);
    __bf16* kh    = (__bf16*)alloc((size_t)BB * HH * PPP * HDD * 2);
    __bf16* vhT   = (__bf16*)alloc((size_t)BB * HH * HDD * PPP * 2);
    float*  logits= (float*) alloc((size_t)BB * HH * LL * PPP * 4);
    __bf16* probs = (__bf16*)alloc((size_t)BB * HH * LL * PPP * 2);
    __bf16* attnb = (__bf16*)alloc((size_t)MTOK * DD * 2);
    __bf16* ffh   = (__bf16*)alloc((size_t)MTOK * NHIDD * 2);
    float*  pred  = (float*) alloc((size_t)BB * PPP * (NOUTT + 1) * 4);

    auto cgrid = [](long n) { return dim3((unsigned)((n + 255) / 256)); };
    auto ntiles = [](int n) { return (unsigned)((n + 127) / 128); };

    // -------- weight transpose-conversion (f32 (K,N) -> bf16 (N,K)) --------
    for (int i = 0; i < NLL; ++i) {
        convT<<<cgrid((long)DD * DD), 256, 0, stream>>>(
            Wq + (size_t)i * DD * DD, wqT + (size_t)i * DD * DD, DD, DD, DD, DD);
        convT<<<cgrid((long)2 * DD * DD), 256, 0, stream>>>(
            Wkv + (size_t)i * DD * 2 * DD, wkvT + (size_t)i * 2 * DD * DD, DD, DD, 2 * DD, 2 * DD);
        convT<<<cgrid((long)DD * DD), 256, 0, stream>>>(
            Wout + (size_t)i * DD * DD, woutT + (size_t)i * DD * DD, DD, DD, DD, DD);
        convT<<<cgrid((long)NHIDD * DD), 256, 0, stream>>>(
            ffW1 + (size_t)i * DD * NHIDD, ffw1T + (size_t)i * NHIDD * DD, DD, DD, NHIDD, NHIDD);
        convT<<<cgrid((long)DD * NHIDD), 256, 0, stream>>>(
            ffW2 + (size_t)i * NHIDD * DD, ffw2T + (size_t)i * DD * NHIDD, NHIDD, NHIDD, DD, DD);
    }
    convT<<<cgrid((long)NHIDD * DD), 256, 0, stream>>>(hW1, hw1T, DD, DD, NHIDD, NHIDD);
    convT<<<cgrid((long)128 * NHIDD), 256, 0, stream>>>(hW2, hw2T, NHIDD, NHIDD, NOUTT + 1, 128);
    convT<<<cgrid((long)DD * KENC), 256, 0, stream>>>(enc_W, encwT, KENC, NFF, DD, DD);

    // -------- preprocessing + encoder --------
    zero_xpad<<<cgrid((long)MTOK * (KENC - NFF)), 256, 0, stream>>>(xp);
    preprocess_kernel<<<dim3(BB * NFF), 256, 0, stream>>>(x_src, xp);

    // xe = xp @ encW  (M=8192, N=1024, K=128) -> qf (f32, reused as xe)
    gemm_bt<false, false, false, false><<<dim3(ntiles(DD), MTOK / 128, 1), 256, 0, stream>>>(
        xp, encwT, nullptr, qf, nullptr, MTOK, DD, KENC, DD, 0, 0, 0, 0, 1);
    enc_epilogue<<<dim3(MTOK), 256, 0, stream>>>(qf, y_src, yenc_W, yenc_b, src);

    // -------- transformer layers --------
    for (int i = 0; i < NLL; ++i) {
        const __bf16* wqT_i   = wqT   + (size_t)i * DD * DD;
        const __bf16* wkvT_i  = wkvT  + (size_t)i * 2 * DD * DD;
        const __bf16* woutT_i = woutT + (size_t)i * DD * DD;
        const __bf16* ffw1T_i = ffw1T + (size_t)i * NHIDD * DD;
        const __bf16* ffw2T_i = ffw2T + (size_t)i * DD * NHIDD;

        // h = LN(src)
        row_ln<<<dim3(MTOK), 256, 0, stream>>>(src, an_g + i * DD, an_b + i * DD, h);

        // Q = h @ Wq   (8192 x 1024 x 1024) -> qf f32
        gemm_bt<false, false, false, false><<<dim3(ntiles(DD), MTOK / 128, 1), 256, 0, stream>>>(
            h, wqT_i, nullptr, qf, nullptr, MTOK, DD, DD, DD, 0, 0, 0, 0, 1);

        // KV = h[:, :P] @ Wkv  (batched over b: 1024 x 2048 x 1024) -> kvf f32
        gemm_bt<false, false, false, false><<<dim3(ntiles(2 * DD), PPP / 128, BB), 256, 0, stream>>>(
            h, wkvT_i, nullptr, kvf, nullptr, PPP, 2 * DD, DD, 2 * DD,
            (long)LL * DD, 0, (long)PPP * 2 * DD, 0, 1);

        // per-head LN of q and k -> head-major bf16
        head_ln<<<dim3(BB * LL * HH), 128, 0, stream>>>(
            qf, (long)DD, LL, qn_g + i * HDD, qn_b + i * HDD, qh);
        head_ln<<<dim3(BB * PPP * HH), 128, 0, stream>>>(
            kvf, (long)2 * DD, PPP, kn_g + i * HDD, kn_b + i * HDD, kh);
        extract_vT<<<cgrid((long)BB * HH * HDD * PPP), 256, 0, stream>>>(kvf, vhT);

        // logits = q @ k^T  (32 batch-heads: 2048 x 1024 x 128) f32
        gemm_bt<false, false, false, false><<<dim3(ntiles(PPP), LL / 128, BB * HH), 256, 0, stream>>>(
            qh, kh, nullptr, logits, nullptr, LL, PPP, HDD, PPP,
            (long)LL * HDD, (long)PPP * HDD, (long)LL * PPP, 0, 1);

        // softmax -> bf16 probs
        softmax_row<<<dim3(BB * HH * LL), 256, 0, stream>>>(logits, probs);

        // attn = probs @ v  (32 bh: 2048 x 128 x 1024) -> merged-head bf16
        gemm_bt<false, false, false, true><<<dim3(ntiles(HDD), LL / 128, BB * HH), 256, 0, stream>>>(
            probs, vhT, nullptr, nullptr, attnb, LL, HDD, PPP, DD,
            (long)LL * PPP, (long)HDD * PPP, (long)LL * DD, (long)HDD, HH);

        // src += attn @ Wout  (8192 x 1024 x 1024, residual)
        gemm_bt<false, false, true, false><<<dim3(ntiles(DD), MTOK / 128, 1), 256, 0, stream>>>(
            attnb, woutT_i, nullptr, src, nullptr, MTOK, DD, DD, DD, 0, 0, 0, 0, 1);

        // h = LN(src); ffh = gelu(h @ ffW1 + b1); src += ffh @ ffW2 + b2
        row_ln<<<dim3(MTOK), 256, 0, stream>>>(src, fn_g + i * DD, fn_b + i * DD, h);
        gemm_bt<true, true, false, true><<<dim3(ntiles(NHIDD), MTOK / 128, 1), 256, 0, stream>>>(
            h, ffw1T_i, ffb1 + (size_t)i * NHIDD, nullptr, ffh,
            MTOK, NHIDD, DD, NHIDD, 0, 0, 0, 0, 1);
        gemm_bt<true, false, true, false><<<dim3(ntiles(DD), MTOK / 128, 1), 256, 0, stream>>>(
            ffh, ffw2T_i, ffb2 + (size_t)i * DD, src, nullptr,
            MTOK, DD, NHIDD, DD, 0, 0, 0, 0, 1);
    }

    // -------- head (only rows l >= P of each batch) --------
    f32_to_bf16<<<cgrid((long)MTOK * DD), 256, 0, stream>>>(src, h, (long)MTOK * DD);

    // ffh = gelu(src[b*L+P : b*L+L] @ hW1 + hb1)   (batched: 1024 x 4096 x 1024)
    gemm_bt<true, true, false, true><<<dim3(ntiles(NHIDD), PPP / 128, BB), 256, 0, stream>>>(
        h + (size_t)PPP * DD, hw1T, hb1, nullptr, ffh,
        PPP, NHIDD, DD, NHIDD, (long)LL * DD, 0, (long)PPP * NHIDD, 0, 1);

    // pred = ffh @ hW2 + hb2   (batched: 1024 x 11 x 4096) f32
    gemm_bt<true, false, false, false><<<dim3(1, PPP / 128, BB), 256, 0, stream>>>(
        ffh, hw2T, hb2, pred, nullptr,
        PPP, NOUTT + 1, NHIDD, NOUTT + 1, (long)PPP * NHIDD, 0, (long)PPP * (NOUTT + 1), 0, 1);

    write_out<<<cgrid((long)PPP * BB * NOUTT), 256, 0, stream>>>(pred, out);
}
